// SparseMLP_43190191129206
// MI455X (gfx1250) — matrix operand order repaired and checked by reading, not compiled, not run on hardware
//
#include <hip/hip_runtime.h>
#include <math.h>

// Problem constants (B=4, S=4096 -> 16384 rows)
#define HD   2048      // hidden dim H
#define ID   8192      // intermediate dim I
#define KTOP 32        // top-k
#define NROWS 16384    // B*S
#define MT   32        // rows per workgroup (two 16-row WMMA tiles)
#define NCHUNK 128     // columns per chunk (8 waves * 16)

typedef __attribute__((ext_vector_type(16))) __bf16       v16bf;
typedef __attribute__((ext_vector_type(8)))  float        v8f;
typedef __attribute__((ext_vector_type(4)))  unsigned int v4u;

union BFU {
    v16bf v;
    unsigned short u[16];
    v4u q[2];          // 2 x 16B = 32B = 16 bf16
};

// f32 -> bf16 round-to-nearest-even
static __device__ __forceinline__ unsigned short f2bf(float f) {
    unsigned u = __float_as_uint(f);
    u += 0x7FFFu + ((u >> 16) & 1u);
    return (unsigned short)(u >> 16);
}
static __device__ __forceinline__ float bf2f(unsigned short h) {
    return __uint_as_float(((unsigned)h) << 16);
}
static __device__ __forceinline__ float gelu_exact(float x) {
    return 0.5f * x * (1.0f + erff(x * 0.70710678118654752f));
}

// ---------------------------------------------------------------------------
// Kernel 0: split W1 (f32) into bf16 hi/lo planes in workspace.
// Pure bandwidth (~134MB @ 23.3TB/s ~ 6us), amortizes the f32->bf16x2 split
// that 512 GEMM workgroups would otherwise each redo from L2.
// ---------------------------------------------------------------------------
__global__ __launch_bounds__(256)
void w1_split_kernel(const float* __restrict__ W1,
                     unsigned short* __restrict__ wh,
                     unsigned short* __restrict__ wl)
{
    const size_t total  = (size_t)ID * HD;
    const size_t stride = (size_t)gridDim.x * blockDim.x * 4;
    for (size_t i = ((size_t)blockIdx.x * blockDim.x + threadIdx.x) * 4;
         i < total; i += stride) {
        float4 f = *(const float4*)(W1 + i);
        float vv[4] = {f.x, f.y, f.z, f.w};
        ushort4 h, l;
        unsigned short hh[4], ll[4];
        #pragma unroll
        for (int j = 0; j < 4; ++j) {
            hh[j] = f2bf(vv[j]);
            ll[j] = f2bf(vv[j] - bf2f(hh[j]));
        }
        h.x = hh[0]; h.y = hh[1]; h.z = hh[2]; h.w = hh[3];
        l.x = ll[0]; l.y = ll[1]; l.z = ll[2]; l.w = ll[3];
        *(ushort4*)(wh + i) = h;
        *(ushort4*)(wl + i) = l;
    }
}

// ---------------------------------------------------------------------------
// Kernel 1: fused GEMM(bf16x3 WMMA) + bias + exact GELU + streaming top-32 +
// emb gather/weighted-sum.  PRECOMP: B fragments loaded pre-split from ws.
// ---------------------------------------------------------------------------
template <bool PRECOMP>
__global__ __launch_bounds__(256)
void sparse_mlp_fused(const float* __restrict__ x,
                      const float* __restrict__ W1,
                      const unsigned short* __restrict__ wh,
                      const unsigned short* __restrict__ wl,
                      const float* __restrict__ b1,
                      const float* __restrict__ emb,
                      float* __restrict__ out)
{
    // LDS: xh[MT][HD] bf16 | xl[MT][HD] bf16 | stage[MT][NCHUNK] f32
    //      topv[MT][KTOP] f32 | topi[MT][KTOP] i32   (total 286720 B < 320KB)
    extern __shared__ unsigned short smem[];
    unsigned short* xh    = smem;
    unsigned short* xl    = xh + MT * HD;
    float*          stage = (float*)(xl + MT * HD);
    float*          topv  = stage + MT * NCHUNK;
    int*            topi  = (int*)(topv + MT * KTOP);

    const int tid  = threadIdx.x;
    const int lane = tid & 31;
    const int wave = tid >> 5;
    const int rowBase = blockIdx.x * MT;

    // ---- Stage x tile -> LDS as bf16 hi/lo planes (bf16x3 split) ----
    for (int i = tid * 4; i < MT * HD; i += 256 * 4) {
        float4 f = *(const float4*)(x + (size_t)rowBase * HD + i);
        float vv[4] = {f.x, f.y, f.z, f.w};
        #pragma unroll
        for (int j = 0; j < 4; ++j) {
            unsigned short h = f2bf(vv[j]);
            xh[i + j] = h;
            xl[i + j] = f2bf(vv[j] - bf2f(h));
        }
    }
    if (tid < MT) {
        #pragma unroll
        for (int k = 0; k < KTOP; ++k) {
            topv[tid * KTOP + k] = -INFINITY;
            topi[tid * KTOP + k] = 0;
        }
    }
    __syncthreads();

    // Per-row streaming top-k state (used by tid < MT only)
    float curMin  = -INFINITY;
    int   minSlot = 0;

    // WMMA operand lane mapping (contiguous k order used by this code, same for A and B):
    // A (16x32 bf16): lane l<16 -> row l, K 0..15 contiguous; lane>=16 -> row l-16, K 16..31
    // B (32x16 bf16): lane l<16 -> col l, K 0..15 contiguous; lane>=16 -> col l-16, K 16..31
    const int aRow  = lane & 15;
    const int aKoff = (lane >> 4) * 16;
    const int bCol  = lane & 15;
    const int bKoff = (lane >> 4) * 16;
    const int cRow  = (lane >> 4) * 8;

    for (int chunk = 0; chunk < ID / NCHUNK; ++chunk) {
        const int colBase = chunk * NCHUNK + wave * 16;
        v8f acc0 = {};   // rows 0..15 of tile
        v8f acc1 = {};   // rows 16..31 of tile

        const float*          wrow  = W1 + (size_t)(colBase + bCol) * HD + bKoff;
        const unsigned short* whrow = PRECOMP ? wh + (size_t)(colBase + bCol) * HD + bKoff : nullptr;
        const unsigned short* wlrow = PRECOMP ? wl + (size_t)(colBase + bCol) * HD + bKoff : nullptr;

        for (int kk = 0; kk < HD; kk += 32) {
            // ---- B fragment: 16 contiguous K-values of this W1 row, hi/lo bf16 ----
            BFU bh, bl;
            if (PRECOMP) {
                bh.q[0] = *(const v4u*)(whrow + kk);
                bh.q[1] = *(const v4u*)(whrow + kk + 8);
                bl.q[0] = *(const v4u*)(wlrow + kk);
                bl.q[1] = *(const v4u*)(wlrow + kk + 8);
            } else {
                const float4* wp = (const float4*)(wrow + kk);
                float4 t0 = wp[0], t1 = wp[1], t2 = wp[2], t3 = wp[3];
                float fb[16] = {t0.x,t0.y,t0.z,t0.w, t1.x,t1.y,t1.z,t1.w,
                                t2.x,t2.y,t2.z,t2.w, t3.x,t3.y,t3.z,t3.w};
                #pragma unroll
                for (int i = 0; i < 16; ++i) {
                    unsigned short h = f2bf(fb[i]);
                    bh.u[i] = h;
                    bl.u[i] = f2bf(fb[i] - bf2f(h));
                }
            }

            // ---- A fragments from LDS (hi/lo, two 16-row tiles) ----
            const int ab0 = aRow * HD + kk + aKoff;        // rows 0..15
            const int ab1 = ab0 + 16 * HD;                 // rows 16..31
            BFU ah0, al0, ah1, al1;
            ah0.q[0] = *(const v4u*)(xh + ab0);
            ah0.q[1] = *(const v4u*)(xh + ab0 + 8);
            al0.q[0] = *(const v4u*)(xl + ab0);
            al0.q[1] = *(const v4u*)(xl + ab0 + 8);
            ah1.q[0] = *(const v4u*)(xh + ab1);
            ah1.q[1] = *(const v4u*)(xh + ab1 + 8);
            al1.q[0] = *(const v4u*)(xl + ab1);
            al1.q[1] = *(const v4u*)(xl + ab1 + 8);

            // bf16x3: AhBh + AlBh + AhBl (drop AlBl), fp32 accumulate
            acc0 = __builtin_amdgcn_wmma_f32_16x16x32_bf16(false, ah0.v, false, bh.v, (short)0, acc0, false, false);
            acc0 = __builtin_amdgcn_wmma_f32_16x16x32_bf16(false, al0.v, false, bh.v, (short)0, acc0, false, false);
            acc0 = __builtin_amdgcn_wmma_f32_16x16x32_bf16(false, ah0.v, false, bl.v, (short)0, acc0, false, false);
            acc1 = __builtin_amdgcn_wmma_f32_16x16x32_bf16(false, ah1.v, false, bh.v, (short)0, acc1, false, false);
            acc1 = __builtin_amdgcn_wmma_f32_16x16x32_bf16(false, al1.v, false, bh.v, (short)0, acc1, false, false);
            acc1 = __builtin_amdgcn_wmma_f32_16x16x32_bf16(false, ah1.v, false, bl.v, (short)0, acc1, false, false);
        }

        // ---- Epilogue: bias + exact GELU -> LDS stage ----
        const float bias = b1[colBase + bCol];
        #pragma unroll
        for (int v = 0; v < 8; ++v) {
            float g0 = gelu_exact(acc0[v] + bias);
            float g1 = gelu_exact(acc1[v] + bias);
            stage[(cRow + v) * NCHUNK + wave * 16 + bCol]      = g0;
            stage[(16 + cRow + v) * NCHUNK + wave * 16 + bCol] = g1;
        }
        __syncthreads();

        // ---- Streaming exact top-32: one lane per row ----
        if (tid < MT) {
            const int r = tid;
            for (int c = 0; c < NCHUNK; ++c) {
                float vv = stage[r * NCHUNK + c];
                if (vv > curMin) {
                    topv[r * KTOP + minSlot] = vv;
                    topi[r * KTOP + minSlot] = chunk * NCHUNK + c;
                    curMin = topv[r * KTOP];
                    minSlot = 0;
                    #pragma unroll
                    for (int k2 = 1; k2 < KTOP; ++k2) {
                        float tv = topv[r * KTOP + k2];
                        if (tv < curMin) { curMin = tv; minSlot = k2; }
                    }
                }
            }
        }
        __syncthreads();
    }

    // ---- Gather + weighted sum: out[row,:] = sum_k G_k * emb[idx_k,:] ----
    for (int r = 0; r < MT; ++r) {
        for (int c = tid; c < HD; c += 256) {
            float acc = 0.f;
            #pragma unroll 8
            for (int k = 0; k < KTOP; ++k) {
                float g  = topv[r * KTOP + k];
                int   id = topi[r * KTOP + k];
                acc += g * emb[(size_t)id * HD + c];
            }
            out[(size_t)(rowBase + r) * HD + c] = acc;
        }
    }
}

extern "C" void kernel_launch(void* const* d_in, const int* in_sizes, int n_in,
                              void* d_out, int out_size, void* d_ws, size_t ws_size,
                              hipStream_t stream) {
    (void)in_sizes; (void)n_in; (void)out_size;
    const float* x   = (const float*)d_in[0];
    const float* W1  = (const float*)d_in[1];
    const float* b1  = (const float*)d_in[2];
    const float* emb = (const float*)d_in[3];
    float* out = (float*)d_out;

    const size_t shmem = (size_t)(2 * MT * HD * 2) + (size_t)MT * NCHUNK * 4
                       + (size_t)MT * KTOP * 4 + (size_t)MT * KTOP * 4;   // 286720 B
    dim3 grid(NROWS / MT);   // 512
    dim3 block(256);

    const size_t planeElems = (size_t)ID * HD;
    const size_t wsNeed     = planeElems * 2 * sizeof(unsigned short);    // 67 MB

    if (ws_size >= wsNeed) {
        unsigned short* wh = (unsigned short*)d_ws;
        unsigned short* wl = wh + planeElems;
        w1_split_kernel<<<dim3(2048), block, 0, stream>>>(W1, wh, wl);
        sparse_mlp_fused<true><<<grid, block, shmem, stream>>>(x, W1, wh, wl, b1, emb, out);
    } else {
        sparse_mlp_fused<false><<<grid, block, shmem, stream>>>(x, W1, nullptr, nullptr, b1, emb, out);
    }
}
